// RNNSeq2Seq_54125177864233
// MI455X (gfx1250) — compile-verified, matrix operand order checked by reading, not run on hardware
//
#include <hip/hip_runtime.h>
#include <hip/hip_bf16.h>
#include <stddef.h>

// ---------------- problem constants (match reference) ----------------
constexpr int VOCAB  = 32000;
constexpr int EMB    = 512;
constexpr int HID    = 1024;
constexpr int S_LEN  = 128;
constexpr int BATCH  = 32;
constexpr int DSTEPS = 48;
constexpr int SOS_TOK = 1;
constexpr int EOS_TOK = 2;

typedef __bf16 bf16_t;
typedef bf16_t bf16x16 __attribute__((ext_vector_type(16)));
typedef float  f32x8   __attribute__((ext_vector_type(8)));

union TileU { bf16x16 v; float4 f4[2]; };

__device__ __forceinline__ float sigmoidf_(float x) { return 1.0f / (1.0f + __expf(-x)); }

// Load a 16x32 bf16 tile (16 rows x 32 K) from row-major memory into the
// CDNA5 WMMA A/B operand layout:
//   lanes 0-15 : row = lane,      K 0..7 in v0-3,  K 16..23 in v4-7
//   lanes 16-31: row = lane-16,   K 8..15 in v0-3, K 24..31 in v4-7
// i.e. per lane: two contiguous 16-byte chunks at +0 and +32 bytes.
__device__ __forceinline__ bf16x16 load_tile16x32(const bf16_t* base, int ld, int lane) {
  const int row  = lane & 15;
  const int half = lane >> 4;
  const float4* p = reinterpret_cast<const float4*>(base + (size_t)row * ld + half * 8);
  TileU u;
  u.f4[0] = p[0];   // 8 bf16
  u.f4[1] = p[2];   // 8 bf16, +16 elements
  return u.v;
}

__device__ __forceinline__ f32x8 wmma_bf16(bf16x16 a, bf16x16 b, f32x8 c) {
  return __builtin_amdgcn_wmma_f32_16x16x32_bf16(
      /*neg_a=*/false, a, /*neg_b=*/false, b,
      /*c_mod=*/(short)0, c, /*reuse_a=*/false, /*reuse_b=*/false);
}

// C/D layout: vgpr i -> row m = mbase + i + 8*(lane>=16), col n = nbase + (lane&15)
__device__ __forceinline__ void store_tile(const f32x8& acc, float bv,
                                           float* __restrict__ C, bf16_t* __restrict__ Cb,
                                           int ldc, int mbase, int n, int lane) {
  const int m0 = mbase + 8 * (lane >> 4);
#pragma unroll
  for (int i = 0; i < 8; ++i) {
    float v = acc[i] + bv;
    size_t idx = (size_t)(m0 + i) * ldc + n;
    if (C)  C[idx]  = v;
    if (Cb) Cb[idx] = (bf16_t)v;
  }
}

// C = A(M,K) * W(N,K)^T + bias, bf16 operands, f32 accumulation.
// Register-blocked: one wave computes a 32x32 output block (2 M-tiles x
// 2 N-tiles, 4 accumulators) so each A/B tile feeds two v_wmma ops:
// 4 WMMAs per 4KB of operand traffic (16 FLOP/B from L2) instead of 8.
// 4 waves per block along N. M and N must be multiples of 32 (true for
// every GEMM in this net). Optional f32 and/or bf16 outputs.
__global__ void wmma_gemm_bias(const bf16_t* __restrict__ A, int lda,
                               const bf16_t* __restrict__ W, int ldw,
                               const float*  __restrict__ bias,
                               float*  __restrict__ C,
                               bf16_t* __restrict__ Cb,
                               int ldc, int N, int K)
{
  const int lane = threadIdx.x & 31;
  const int wave = threadIdx.x >> 5;
  const int nb   = blockIdx.x * 4 + wave;   // 32-wide N block
  const int mb   = blockIdx.y;              // 32-wide M block
  if (nb * 32 >= N) return;  // wave-uniform; never taken with our launch shapes

  const bf16_t* A0 = A + (size_t)(mb * 32 +  0) * lda;
  const bf16_t* A1 = A + (size_t)(mb * 32 + 16) * lda;
  const bf16_t* W0 = W + (size_t)(nb * 32 +  0) * ldw;
  const bf16_t* W1 = W + (size_t)(nb * 32 + 16) * ldw;

  f32x8 acc00 = {0.f,0.f,0.f,0.f,0.f,0.f,0.f,0.f};
  f32x8 acc01 = acc00, acc10 = acc00, acc11 = acc00;

  for (int k = 0; k < K; k += 32) {
    bf16x16 a0 = load_tile16x32(A0 + k, lda, lane);
    bf16x16 a1 = load_tile16x32(A1 + k, lda, lane);
    bf16x16 b0 = load_tile16x32(W0 + k, ldw, lane);
    bf16x16 b1 = load_tile16x32(W1 + k, ldw, lane);
    __builtin_prefetch(W0 + k + 256, 0, 3);  // warm near cache for weight stream
    __builtin_prefetch(W1 + k + 256, 0, 3);
    acc00 = wmma_bf16(a0, b0, acc00);
    acc01 = wmma_bf16(a0, b1, acc01);
    acc10 = wmma_bf16(a1, b0, acc10);
    acc11 = wmma_bf16(a1, b1, acc11);
  }

  const int n0  = nb * 32 + (lane & 15);
  const int n1  = n0 + 16;
  const float bv0 = bias ? bias[n0] : 0.0f;
  const float bv1 = bias ? bias[n1] : 0.0f;
  const int m0 = mb * 32;
  store_tile(acc00, bv0, C, Cb, ldc, m0,      n0, lane);
  store_tile(acc01, bv1, C, Cb, ldc, m0,      n1, lane);
  store_tile(acc10, bv0, C, Cb, ldc, m0 + 16, n0, lane);
  store_tile(acc11, bv1, C, Cb, ldc, m0 + 16, n1, lane);
}

// ---------------- elementwise / gather kernels ----------------

__global__ void f32_to_bf16_k(const float* __restrict__ s, bf16_t* __restrict__ d, size_t n) {
  size_t i = (size_t)blockIdx.x * blockDim.x + threadIdx.x;
  size_t stride = (size_t)gridDim.x * blockDim.x;
  for (; i < n; i += stride) d[i] = (bf16_t)s[i];
}

__global__ void zero_u32(unsigned* __restrict__ p, size_t n) {
  size_t i = (size_t)blockIdx.x * blockDim.x + threadIdx.x;
  size_t stride = (size_t)gridDim.x * blockDim.x;
  for (; i < n; i += stride) p[i] = 0u;
}

__global__ void embed_seq(const int* __restrict__ seq, const float* __restrict__ emb,
                          bf16_t* __restrict__ x0) {
  const int row = blockIdx.x;             // row = s*B + b, matches x[s,b]
  const int tk  = seq[row];
  for (int e = threadIdx.x; e < EMB; e += blockDim.x)
    x0[(size_t)row * EMB + e] = (bf16_t)emb[(size_t)tk * EMB + e];
}

// Encoder GRU pointwise for both directions (z = dir): consumes precomputed
// gi (all timesteps) and this step's gh; updates f32 h state, the bf16
// concat-hidden (B,2H) used as the next recurrent A operand, and the
// layer output y (S*B, 2H) in bf16 for the next layer's batched GEMM.
__global__ void gru_pointwise_enc(const float* __restrict__ gi_f, const float* __restrict__ gi_b,
                                  const float* __restrict__ gh_f, const float* __restrict__ gh_b,
                                  float*  __restrict__ h,      // (2, B, H) f32 state
                                  bf16_t* __restrict__ hcat,   // (B, 2H) bf16
                                  bf16_t* __restrict__ y,      // (S*B, 2H) bf16
                                  int t)
{
  const int dir = blockIdx.z;
  const int b   = blockIdx.y;
  const int j   = blockIdx.x * blockDim.x + threadIdx.x;
  const int tt  = dir ? (S_LEN - 1 - t) : t;
  const float* gi = (dir ? gi_b : gi_f) + (size_t)(tt * BATCH + b) * (3 * HID);
  const float* gh = (dir ? gh_b : gh_f) + (size_t)b * (3 * HID);
  float* hp = h + ((size_t)dir * BATCH + b) * HID;

  float hv = hp[j];
  float r  = sigmoidf_(gi[j]           + gh[j]);
  float z  = sigmoidf_(gi[HID + j]     + gh[HID + j]);
  float nn = tanhf   (gi[2 * HID + j]  + r * gh[2 * HID + j]);
  float h2 = (1.0f - z) * nn + z * hv;

  hp[j] = h2;
  bf16_t hb = (bf16_t)h2;
  hcat[(size_t)b * (2 * HID) + dir * HID + j] = hb;
  y[((size_t)tt * BATCH + b) * (2 * HID) + dir * HID + j] = hb;
}

__global__ void gru_pointwise_dec(const float* __restrict__ gi, const float* __restrict__ gh,
                                  const float* __restrict__ hprev,
                                  float* __restrict__ hnew, bf16_t* __restrict__ hnew_b16)
{
  const int b = blockIdx.y;
  const int j = blockIdx.x * blockDim.x + threadIdx.x;
  const float* gip = gi + (size_t)b * (3 * HID);
  const float* ghp = gh + (size_t)b * (3 * HID);
  float hv = hprev[(size_t)b * HID + j];
  float r  = sigmoidf_(gip[j]          + ghp[j]);
  float z  = sigmoidf_(gip[HID + j]    + ghp[HID + j]);
  float nn = tanhf   (gip[2 * HID + j] + r * ghp[2 * HID + j]);
  float h2 = (1.0f - z) * nn + z * hv;
  hnew[(size_t)b * HID + j]     = h2;
  hnew_b16[(size_t)b * HID + j] = (bf16_t)h2;
}

__global__ void init_dec(const float* __restrict__ emb, bf16_t* __restrict__ dec_in_b16,
                         int* __restrict__ alive)
{
  const int b = blockIdx.y;
  const int e = blockIdx.x * blockDim.x + threadIdx.x;  // grid (E/256, B)
  dec_in_b16[(size_t)b * EMB + e] = (bf16_t)emb[(size_t)SOS_TOK * EMB + e];
  if (e == 0 && blockIdx.x == 0) alive[b] = 1;
}

// One block per batch row: rowmax+argmax (first-occurrence), sum of exp,
// write dist (or the PAD one-hot for finished rows), publish argmax token.
__global__ void softmax_argmax_write(const float* __restrict__ logits,
                                     float* __restrict__ out,
                                     int* __restrict__ tok,
                                     const int* __restrict__ alive)
{
  const int b   = blockIdx.x;
  const int tid = threadIdx.x;
  const float* row = logits + (size_t)b * VOCAB;
  __shared__ float sv[256];
  __shared__ int   si[256];

  float best = -3.4e38f; int bidx = 0x7fffffff;
  for (int v = tid; v < VOCAB; v += 256) {
    float x = row[v];
    if (x > best) { best = x; bidx = v; }   // ascending v -> lowest index kept
  }
  sv[tid] = best; si[tid] = bidx;
  __syncthreads();
  for (int s = 128; s > 0; s >>= 1) {
    if (tid < s) {
      float ov = sv[tid + s]; int oi = si[tid + s];
      if (ov > sv[tid] || (ov == sv[tid] && oi < si[tid])) { sv[tid] = ov; si[tid] = oi; }
    }
    __syncthreads();
  }
  const float m   = sv[0];
  const int   arg = si[0];
  __syncthreads();

  float sum = 0.f;
  for (int v = tid; v < VOCAB; v += 256) sum += __expf(row[v] - m);
  sv[tid] = sum;
  __syncthreads();
  for (int s = 128; s > 0; s >>= 1) { if (tid < s) sv[tid] += sv[tid + s]; __syncthreads(); }
  const float inv  = 1.0f / sv[0];
  const int   live = alive[b];

  float* orow = out + (size_t)b * VOCAB;
  for (int v = tid; v < VOCAB; v += 256) {
    float d = live ? __expf(row[v] - m) * inv : (v == 0 ? 1.0f : 0.0f);
    orow[v] = d;
  }
  if (tid == 0) tok[b] = arg;
}

// Masked state update (uses pre-update alive mask, then advances it).
__global__ void dec_update(const int* __restrict__ tok, int* __restrict__ alive,
                           const float* __restrict__ hnew,
                           float* __restrict__ dec_h, bf16_t* __restrict__ dec_h_b16,
                           const float* __restrict__ emb, bf16_t* __restrict__ dec_in_b16)
{
  const int b    = blockIdx.x;
  const int tid  = threadIdx.x;     // HID threads
  const int live = alive[b];
  const int t    = tok[b];
  if (live) {
    float hv = hnew[(size_t)b * HID + tid];
    dec_h[(size_t)b * HID + tid]     = hv;
    dec_h_b16[(size_t)b * HID + tid] = (bf16_t)hv;
    if (tid < EMB) dec_in_b16[(size_t)b * EMB + tid] = (bf16_t)emb[(size_t)t * EMB + tid];
  }
  __syncthreads();
  if (tid == 0) alive[b] = live && (t != EOS_TOK);
}

// -------------------------------------------------------------------------

extern "C" void kernel_launch(void* const* d_in, const int* in_sizes, int n_in,
                              void* d_out, int out_size, void* d_ws, size_t ws_size,
                              hipStream_t stream) {
  (void)in_sizes; (void)n_in; (void)out_size; (void)ws_size;

  // Input order: top-level dict insertion order; nested params flattened in
  // JAX tree order (dict keys sorted by code point, capitals < lowercase):
  //  0 input_seq  1 input_lengths  2 decoding_steps
  //  3 b2u_W  4 b2u_b  5 dec_Wh  6 dec_Wi  7 dec_bh  8 dec_bi
  //  9 dec_in_W 10 dec_in_b 11 dec_out_W 12 dec_out_b 13 emb
  //  14.. enc[l] in order: Wh_b, Wh_f, Wi_b, Wi_f, bh_b, bh_f, bi_b, bi_f
  const int*   input_seq = (const int*)  d_in[0];
  const float* b2u_W     = (const float*)d_in[3];
  const float* b2u_b     = (const float*)d_in[4];
  const float* dec_Wh    = (const float*)d_in[5];
  const float* dec_Wi    = (const float*)d_in[6];
  const float* dec_bh    = (const float*)d_in[7];
  const float* dec_bi    = (const float*)d_in[8];
  const float* dec_in_W  = (const float*)d_in[9];
  const float* dec_in_b  = (const float*)d_in[10];
  const float* dec_out_W = (const float*)d_in[11];
  const float* dec_out_b = (const float*)d_in[12];
  const float* emb       = (const float*)d_in[13];
  const float* encp[2][8];
  for (int l = 0; l < 2; ++l)
    for (int i = 0; i < 8; ++i)
      encp[l][i] = (const float*)d_in[14 + l * 8 + i];
  // encp[l]: 0 Wh_b, 1 Wh_f, 2 Wi_b, 3 Wi_f, 4 bh_b, 5 bh_f, 6 bi_b, 7 bi_f

  float* out_f = (float*)d_out;

  // ---- workspace allocator (256B aligned) ----
  char*  wsb = (char*)d_ws;
  size_t off = 0;
  auto alloc = [&](size_t bytes) -> void* {
    void* p = wsb + off;
    off = (off + bytes + 255) & ~(size_t)255;
    return p;
  };
  auto cvt = [&](const float* s, size_t n) -> bf16_t* {
    bf16_t* d = (bf16_t*)alloc(n * sizeof(bf16_t));
    unsigned g = (unsigned)(((n + 255) / 256) < 2048 ? ((n + 255) / 256) : 2048);
    f32_to_bf16_k<<<g, 256, 0, stream>>>(s, d, n);
    return d;
  };
  auto gemm = [&](const bf16_t* A, int lda, const bf16_t* W, int ldw,
                  const float* bias, float* C, bf16_t* Cb, int ldc,
                  int M, int N, int K) {
    dim3 blk(128);
    dim3 grd((unsigned)((N / 32 + 3) / 4), (unsigned)(M / 32));
    wmma_gemm_bias<<<grd, blk, 0, stream>>>(A, lda, W, ldw, bias, C, Cb, ldc, N, K);
  };

  // ---- one-time weight conversion f32 -> bf16 (L2-resident working set) ----
  bf16_t* w_b2u    = cvt(b2u_W,     (size_t)HID * 2 * HID);
  bf16_t* w_decWh  = cvt(dec_Wh,    (size_t)3 * HID * HID);
  bf16_t* w_decWi  = cvt(dec_Wi,    (size_t)3 * HID * HID);
  bf16_t* w_decin  = cvt(dec_in_W,  (size_t)HID * EMB);
  bf16_t* w_decout = cvt(dec_out_W, (size_t)VOCAB * HID);
  bf16_t* w_Wh[2][2]; // [layer][dir] 0=f 1=b
  bf16_t* w_Wi[2][2];
  for (int l = 0; l < 2; ++l) {
    const size_t wiN = (size_t)3 * HID * (l == 0 ? EMB : 2 * HID);
    w_Wh[l][0] = cvt(encp[l][1], (size_t)3 * HID * HID);
    w_Wh[l][1] = cvt(encp[l][0], (size_t)3 * HID * HID);
    w_Wi[l][0] = cvt(encp[l][3], wiN);
    w_Wi[l][1] = cvt(encp[l][2], wiN);
  }

  // ---- activations ----
  bf16_t* xbuf0 = (bf16_t*)alloc((size_t)S_LEN * BATCH * 2 * HID * sizeof(bf16_t));
  bf16_t* xbuf1 = (bf16_t*)alloc((size_t)S_LEN * BATCH * 2 * HID * sizeof(bf16_t));
  float*  gi_f  = (float*)alloc((size_t)S_LEN * BATCH * 3 * HID * sizeof(float));
  float*  gi_b  = (float*)alloc((size_t)S_LEN * BATCH * 3 * HID * sizeof(float));
  float*  gh_f  = (float*)alloc((size_t)BATCH * 3 * HID * sizeof(float));
  float*  gh_b  = (float*)alloc((size_t)BATCH * 3 * HID * sizeof(float));
  float*  h_st  = (float*)alloc((size_t)2 * BATCH * HID * sizeof(float));   // (dir,B,H)
  bf16_t* hcat  = (bf16_t*)alloc((size_t)BATCH * 2 * HID * sizeof(bf16_t)); // (B,2H)

  float*  xin_f32 = (float*)alloc((size_t)BATCH * HID * sizeof(float));
  bf16_t* xin_b16 = (bf16_t*)alloc((size_t)BATCH * HID * sizeof(bf16_t));
  float*  dgi     = (float*)alloc((size_t)BATCH * 3 * HID * sizeof(float));
  float*  dgh     = (float*)alloc((size_t)BATCH * 3 * HID * sizeof(float));
  float*  dec_h   = (float*)alloc((size_t)BATCH * HID * sizeof(float));
  bf16_t* dec_hb  = (bf16_t*)alloc((size_t)BATCH * HID * sizeof(bf16_t));
  float*  hnew    = (float*)alloc((size_t)BATCH * HID * sizeof(float));
  bf16_t* hnewb   = (bf16_t*)alloc((size_t)BATCH * HID * sizeof(bf16_t));
  bf16_t* dec_inb = (bf16_t*)alloc((size_t)BATCH * EMB * sizeof(bf16_t));
  float*  logits  = (float*)alloc((size_t)BATCH * VOCAB * sizeof(float));
  int*    tok     = (int*)alloc(BATCH * sizeof(int));
  int*    alive   = (int*)alloc(BATCH * sizeof(int));

  // ---- encoder ----
  embed_seq<<<S_LEN * BATCH, 256, 0, stream>>>(input_seq, emb, xbuf0); // (S*B, E) bf16

  bf16_t* xa  = xbuf0;
  int     din = EMB;
  for (int l = 0; l < 2; ++l) {
    // hoisted input GEMMs for all timesteps, both directions
    gemm(xa, din, w_Wi[l][0], din, encp[l][7], gi_f, nullptr, 3 * HID,
         S_LEN * BATCH, 3 * HID, din);   // bi_f
    gemm(xa, din, w_Wi[l][1], din, encp[l][6], gi_b, nullptr, 3 * HID,
         S_LEN * BATCH, 3 * HID, din);   // bi_b

    zero_u32<<<256, 256, 0, stream>>>((unsigned*)h_st, (size_t)2 * BATCH * HID);
    zero_u32<<<128, 256, 0, stream>>>((unsigned*)hcat, (size_t)BATCH * 2 * HID / 2);

    bf16_t* y = (l == 0) ? xbuf1 : xbuf0;
    for (int t = 0; t < S_LEN; ++t) {
      gemm(hcat,       2 * HID, w_Wh[l][0], HID, encp[l][5], gh_f, nullptr, 3 * HID,
           BATCH, 3 * HID, HID);   // bh_f
      gemm(hcat + HID, 2 * HID, w_Wh[l][1], HID, encp[l][4], gh_b, nullptr, 3 * HID,
           BATCH, 3 * HID, HID);   // bh_b
      gru_pointwise_enc<<<dim3(HID / 256, BATCH, 2), 256, 0, stream>>>(
          gi_f, gi_b, gh_f, gh_b, h_st, hcat, y, t);
    }
    xa  = y;
    din = 2 * HID;
  }
  // hcat now holds enc_hidden = concat(hf, hb) of the last layer, in bf16.

  // ---- decoder init ----
  gemm(hcat, 2 * HID, w_b2u, 2 * HID, b2u_b, dec_h, dec_hb, HID,
       BATCH, HID, 2 * HID);
  init_dec<<<dim3(EMB / 256, BATCH), 256, 0, stream>>>(emb, dec_inb, alive);

  // ---- decoder loop ----
  for (int st = 0; st < DSTEPS; ++st) {
    gemm(dec_inb, EMB, w_decin, EMB, dec_in_b, xin_f32, xin_b16, HID,
         BATCH, HID, EMB);
    gemm(xin_b16, HID, w_decWi, HID, dec_bi, dgi, nullptr, 3 * HID,
         BATCH, 3 * HID, HID);
    gemm(dec_hb, HID, w_decWh, HID, dec_bh, dgh, nullptr, 3 * HID,
         BATCH, 3 * HID, HID);
    gru_pointwise_dec<<<dim3(HID / 256, BATCH), 256, 0, stream>>>(
        dgi, dgh, dec_h, hnew, hnewb);
    gemm(hnewb, HID, w_decout, HID, dec_out_b, logits, nullptr, VOCAB,
         BATCH, VOCAB, HID);
    softmax_argmax_write<<<BATCH, 256, 0, stream>>>(
        logits, out_f + (size_t)st * BATCH * VOCAB, tok, alive);
    dec_update<<<BATCH, HID, 0, stream>>>(
        tok, alive, hnew, dec_h, dec_hb, emb, dec_inb);
  }

  // final_lengths = zeros (int32 zero bits)
  zero_u32<<<1, 32, 0, stream>>>(
      (unsigned*)(out_f + (size_t)DSTEPS * BATCH * VOCAB), (size_t)BATCH);
}
